// HandEncoderTransformerBiasGlobal_12816182411334
// MI455X (gfx1250) — matrix operand order repaired and checked by reading, not compiled, not run on hardware
//
#include <hip/hip_runtime.h>
#include <cstdint>

// ---------------- problem constants ----------------
#define B_    64
#define N_    256
#define E_    2048
#define D_    256
#define H_    8
#define DH_   32
#define DFF_  512
#define HP_   4
#define DHP_  64
#define OUT_  512
#define NF_   10
#define INPE_PAD 128   // 95 real features padded with zeros to a K multiple of 32

typedef __attribute__((ext_vector_type(16))) _Float16 v16h;
typedef __attribute__((ext_vector_type(8)))  float    v8f;
typedef _Float16 h16_t;

// Build a v16h from two 16-byte chunks (matches CDNA5 WMMA A/B lane layouts:
// halves e=0..7 from p0, e=8..15 from p1).
__device__ inline v16h ld_v16h(const h16_t* p0, const h16_t* p1) {
  v16h r;
  uint4* u = reinterpret_cast<uint4*>(&r);
  u[0] = *reinterpret_cast<const uint4*>(p0);
  u[1] = *reinterpret_cast<const uint4*>(p1);
  return r;
}

__device__ inline float act_eval(float v, int act) {
  if (act == 1) {                       // silu
    return v / (1.0f + __expf(-v));
  }
  if (act == 2) {                       // gelu (tanh approx, jax default)
    float t = tanhf(0.7978845608028654f * (v + 0.044715f * v * v * v));
    return 0.5f * v * (1.0f + t);
  }
  return v;
}

// ---------------- weight pre-pack into WMMA-B tiled layout ----------------
// out[((kt*NT + nt)*32 + lane)*16 + e] = W[kt*32 + (lane>>4)*16 + e][nt*16 + (lane&15)]
__global__ __launch_bounds__(256) void pack_w_kernel(
    const float* __restrict__ W, h16_t* __restrict__ out,
    int Ksrc, int Kpad, int Nc) {
  int lane = threadIdx.x & 31;
  int tile = blockIdx.x * 8 + (threadIdx.x >> 5);
  int NT = Nc >> 4;
  int tiles = (Kpad >> 5) * NT;
  if (tile >= tiles) return;
  int kt = tile / NT, nt = tile % NT;
  int col = nt * 16 + (lane & 15);
  int kb  = kt * 32 + ((lane >> 4) * 16);
  h16_t* o = out + ((size_t)tile * 32 + lane) * 16;
  #pragma unroll
  for (int e = 0; e < 16; ++e) {
    int k = kb + e;
    o[e] = (k < Ksrc) ? (h16_t)W[(size_t)k * Nc + col] : (h16_t)0.0f;
  }
}

// ---------------- generic WMMA GEMM: out = act(A@W + bias [+ res]) ----------------
// A: M x K row-major f16. Wpk: tiled f16 (see pack_w_kernel). 8 waves/block,
// each wave owns a 16x64 output tile (4 accumulators), K stepped by 32.
__global__ __launch_bounds__(256) void gemm_wmma_kernel(
    const h16_t* __restrict__ A, const h16_t* __restrict__ Wpk,
    const float* __restrict__ bias, const float* res,
    float* out32, h16_t* out16,
    int M, int K, int Nc, int act) {
  const int lane = threadIdx.x & 31;
  const int wv   = threadIdx.x >> 5;
  const int rt   = blockIdx.x * 8 + wv;
  if (rt * 16 >= M) return;
  const int ng = blockIdx.y;           // 64-wide column group
  const int NT = Nc >> 4;

  v8f zero = {};
  v8f acc[4];
  acc[0] = zero; acc[1] = zero; acc[2] = zero; acc[3] = zero;

  const int row = rt * 16 + (lane & 15);
  const h16_t* arow = A + (size_t)row * K + ((lane >> 4) * 16);
  const int ksteps = K >> 5;

  for (int kt = 0; kt < ksteps; ++kt) {
    v16h a = ld_v16h(arow + kt * 32, arow + kt * 32 + 8);
    const h16_t* bb = Wpk + (((size_t)kt * NT + ng * 4) * 32 + lane) * 16;
    v16h b0 = ld_v16h(bb,        bb + 8);
    v16h b1 = ld_v16h(bb + 512,  bb + 520);
    v16h b2 = ld_v16h(bb + 1024, bb + 1032);
    v16h b3 = ld_v16h(bb + 1536, bb + 1544);
    acc[0] = __builtin_amdgcn_wmma_f32_16x16x32_f16(false, a, false, b0, (short)0, acc[0], false, false);
    acc[1] = __builtin_amdgcn_wmma_f32_16x16x32_f16(false, a, false, b1, (short)0, acc[1], false, false);
    acc[2] = __builtin_amdgcn_wmma_f32_16x16x32_f16(false, a, false, b2, (short)0, acc[2], false, false);
    acc[3] = __builtin_amdgcn_wmma_f32_16x16x32_f16(false, a, false, b3, (short)0, acc[3], false, false);
  }

  const int r0 = rt * 16 + ((lane >> 4) * 8);
  #pragma unroll
  for (int t = 0; t < 4; ++t) {
    int col = (ng * 4 + t) * 16 + (lane & 15);
    float bc = bias ? bias[col] : 0.0f;
    #pragma unroll
    for (int r = 0; r < 8; ++r) {
      size_t idx = (size_t)(r0 + r) * Nc + col;
      float v = acc[t][r] + bc;
      if (res) v += res[idx];
      v = act_eval(v, act);
      if (out32) out32[idx] = v;
      if (out16) out16[idx] = (h16_t)v;
    }
  }
}

// ---------------- LayerNorm (one wave per row) ----------------
__global__ __launch_bounds__(256) void ln_kernel(
    const float* __restrict__ in, const float* __restrict__ g, const float* __restrict__ bta,
    float* out32, h16_t* out16, int M, int C, int act) {
  int lane = threadIdx.x & 31;
  int row  = blockIdx.x * 8 + (threadIdx.x >> 5);
  if (row >= M) return;
  const float* x = in + (size_t)row * C;
  float s = 0.f, ss = 0.f;
  for (int c = lane; c < C; c += 32) { float v = x[c]; s += v; ss += v * v; }
  #pragma unroll
  for (int o = 16; o > 0; o >>= 1) { s += __shfl_xor(s, o, 32); ss += __shfl_xor(ss, o, 32); }
  float mean = s / C;
  float var  = ss / C - mean * mean;
  float rstd = rsqrtf(var + 1e-5f);
  for (int c = lane; c < C; c += 32) {
    float v = (x[c] - mean) * rstd * g[c] + bta[c];
    v = act_eval(v, act);
    if (out32) out32[(size_t)row * C + c] = v;
    if (out16) out16[(size_t)row * C + c] = (h16_t)v;
  }
}

// ---------------- feature build: xyz fourier PE + embeddings, padded to 128 ----------------
__global__ __launch_bounds__(256) void feats_kernel(
    const float* __restrict__ xyz, const int* __restrict__ fid, const int* __restrict__ jid,
    const float* __restrict__ femb, const float* __restrict__ jemb, h16_t* __restrict__ feats) {
  int t = blockIdx.x * blockDim.x + threadIdx.x;
  if (t >= B_ * N_) return;
  const float* p = xyz + (size_t)t * 3;
  h16_t* o = feats + (size_t)t * INPE_PAD;
  float c3[3] = {p[0], p[1], p[2]};
  o[0] = (h16_t)c3[0]; o[1] = (h16_t)c3[1]; o[2] = (h16_t)c3[2];
  for (int c = 0; c < 3; ++c) {
    #pragma unroll
    for (int f = 0; f < NF_; ++f) {
      float ang = c3[c] * (float)(1 << f);
      o[3  + c * NF_ + f] = (h16_t)sinf(ang);
      o[33 + c * NF_ + f] = (h16_t)cosf(ang);
    }
  }
  int fi = fid[t], ji = jid[t];
  #pragma unroll
  for (int k = 0; k < 16; ++k) o[63 + k] = (h16_t)femb[fi * 16 + k];
  #pragma unroll
  for (int k = 0; k < 16; ++k) o[79 + k] = (h16_t)jemb[ji * 16 + k];
  for (int k = 95; k < INPE_PAD; ++k) o[k] = (h16_t)0.0f;
}

// ---------------- edge MLP (K=10 -> scalar) ----------------
__global__ __launch_bounds__(256) void edge_mlp_kernel(
    const float* __restrict__ xyz, const int* __restrict__ eidx, const int* __restrict__ etype,
    const float* __restrict__ erl, const float* __restrict__ esemb,
    const float* __restrict__ w1, const float* __restrict__ b1,
    const float* __restrict__ w2, const float* __restrict__ b2,
    float* __restrict__ eb) {
  int t = blockIdx.x * blockDim.x + threadIdx.x;
  if (t >= B_ * E_) return;
  int b = t / E_, e = t % E_;
  int i = eidx[e], j = eidx[E_ + e];
  float dx = xyz[((size_t)b * N_ + i) * 3 + 0] - xyz[((size_t)b * N_ + j) * 3 + 0];
  float dy = xyz[((size_t)b * N_ + i) * 3 + 1] - xyz[((size_t)b * N_ + j) * 3 + 1];
  float dz = xyz[((size_t)b * N_ + i) * 3 + 2] - xyz[((size_t)b * N_ + j) * 3 + 2];
  float dist2 = dx * dx + dy * dy + dz * dz;
  float dist  = sqrtf(dist2 + 1e-9f);
  float rest  = erl[e];
  float delta = (dist - rest) / (rest + 1e-9f);
  float f[10];
  f[0] = dist2; f[1] = delta;
  int et = etype[e];
  #pragma unroll
  for (int k = 0; k < 8; ++k) f[2 + k] = esemb[et * 8 + k];
  float out = b2[0];
  for (int o = 0; o < 32; ++o) {
    float a = b1[o];
    #pragma unroll
    for (int k = 0; k < 10; ++k) a += f[k] * w1[k * 32 + o];
    a = a / (1.0f + __expf(-a));    // silu
    out += a * w2[o];
  }
  eb[t] = out;
}

__global__ void bias_fill_kernel(float* bias, const float* neb, int total) {
  int t = blockIdx.x * blockDim.x + threadIdx.x;
  if (t < total) bias[t] = neb[0];
}

__global__ void bias_scatter_kernel(float* bias, const float* __restrict__ eb,
                                    const int* __restrict__ eidx) {
  int t = blockIdx.x * blockDim.x + threadIdx.x;
  if (t >= B_ * E_) return;
  int b = t / E_, e = t % E_;
  float w = eb[t];
  int i = eidx[e], j = eidx[E_ + e];
  bias[((size_t)b * N_ + i) * N_ + j] = w;
  bias[((size_t)b * N_ + j) * N_ + i] = w;
}

// ---------------- pack V into WMMA-B tiled layout per (b,h) ----------------
// tile = ((b*H + h)*8 + kt)*2 + nt ; value = V[b, kt*32+(lane>>4)*16+e, h*32+nt*16+(lane&15)]
__global__ __launch_bounds__(256) void pack_v_kernel(const h16_t* __restrict__ v16,
                                                     h16_t* __restrict__ vpk) {
  int lane = threadIdx.x & 31;
  int tile = blockIdx.x * 8 + (threadIdx.x >> 5);
  if (tile >= B_ * H_ * 16) return;
  int nt = tile & 1, kt = (tile >> 1) & 7, h = (tile >> 4) & 7, b = tile >> 7;
  int col  = h * DH_ + nt * 16 + (lane & 15);
  int tok0 = kt * 32 + ((lane >> 4) * 16);
  h16_t* o = vpk + ((size_t)tile * 32 + lane) * 16;
  #pragma unroll
  for (int e = 0; e < 16; ++e)
    o[e] = v16[((size_t)(b * N_ + tok0 + e)) * D_ + col];
}

// ---------------- fused attention: scores->bias+softmax->attn@V, all in LDS ----------------
// 2 waves/block; each wave owns 16 query rows of one (b,h).
__global__ __launch_bounds__(64) void attn_kernel(
    const h16_t* __restrict__ Q, const h16_t* __restrict__ Km,
    const h16_t* __restrict__ Vpk, const float* __restrict__ bias,
    h16_t* __restrict__ O) {
  __shared__ __align__(16) float  s_sc[2][16][N_];  // 32 KB
  __shared__ __align__(16) h16_t  s_at[2][16][N_];  // 16 KB
  const int lane = threadIdx.x & 31;
  const int wv   = threadIdx.x >> 5;
  int id  = blockIdx.x;
  int qt2 = id & 7;
  int h   = (id >> 3) & 7;
  int b   = id >> 6;
  int n0  = (qt2 * 2 + wv) * 16;

  // --- scores: Q(16x32) x K^T(32x16) per 16-key tile ---
  {
    int row = n0 + (lane & 15);
    const h16_t* qp = Q + ((size_t)(b * N_ + row)) * D_ + h * DH_ + ((lane >> 4) * 16);
    v16h a = ld_v16h(qp, qp + 8);
    for (int kt = 0; kt < 16; ++kt) {
      int nk = kt * 16 + (lane & 15);
      const h16_t* kp = Km + ((size_t)(b * N_ + nk)) * D_ + h * DH_ + ((lane >> 4) * 16);
      v16h bm = ld_v16h(kp, kp + 8);
      v8f c = {};
      c = __builtin_amdgcn_wmma_f32_16x16x32_f16(false, a, false, bm, (short)0, c, false, false);
      int rr = (lane >> 4) * 8;
      int cc = kt * 16 + (lane & 15);
      #pragma unroll
      for (int r = 0; r < 8; ++r) s_sc[wv][rr + r][cc] = c[r];
    }
  }
  __syncthreads();

  // --- scale + bias + row softmax; result f16 in s_at ---
  {
    int row = lane & 15;
    int seg = lane >> 4;                 // each row handled by lane pair (l, l+16)
    const float scale = 0.17677669529663687f;  // 1/sqrt(32)
    const float* brow = bias + ((size_t)b * N_ + (n0 + row)) * N_ + seg * 128;
    float m = -3.4e38f;
    for (int c = 0; c < 128; ++c) {
      float v = s_sc[wv][row][seg * 128 + c] * scale + brow[c];
      s_sc[wv][row][seg * 128 + c] = v;
      m = fmaxf(m, v);
    }
    m = fmaxf(m, __shfl_xor(m, 16, 32));
    float ssum = 0.f;
    for (int c = 0; c < 128; ++c) {
      float ev = __expf(s_sc[wv][row][seg * 128 + c] - m);
      s_sc[wv][row][seg * 128 + c] = ev;
      ssum += ev;
    }
    ssum += __shfl_xor(ssum, 16, 32);
    float inv = 1.0f / ssum;
    for (int c = 0; c < 128; ++c)
      s_at[wv][row][seg * 128 + c] = (h16_t)(s_sc[wv][row][seg * 128 + c] * inv);
  }
  __syncthreads();

  // --- o = attn(16x256) @ V(256x32) ---
  {
    v8f a0 = {}, a1 = {};
    int rowA = lane & 15;
    for (int kt = 0; kt < 8; ++kt) {
      int kb = kt * 32 + ((lane >> 4) * 16);
      v16h a = ld_v16h(&s_at[wv][rowA][kb], &s_at[wv][rowA][kb + 8]);
      int tbase = ((b * H_ + h) * 8 + kt) * 2;
      const h16_t* v0 = Vpk + ((size_t)(tbase + 0) * 32 + lane) * 16;
      const h16_t* v1 = Vpk + ((size_t)(tbase + 1) * 32 + lane) * 16;
      v16h b0 = ld_v16h(v0, v0 + 8);
      v16h b1 = ld_v16h(v1, v1 + 8);
      a0 = __builtin_amdgcn_wmma_f32_16x16x32_f16(false, a, false, b0, (short)0, a0, false, false);
      a1 = __builtin_amdgcn_wmma_f32_16x16x32_f16(false, a, false, b1, (short)0, a1, false, false);
    }
    int r0 = n0 + ((lane >> 4) * 8);
    #pragma unroll
    for (int r = 0; r < 8; ++r) {
      size_t base = ((size_t)(b * N_ + r0 + r)) * D_ + h * DH_ + (lane & 15);
      O[base]      = (h16_t)a0[r];
      O[base + 16] = (h16_t)a1[r];
    }
  }
}

// ---------------- PMA pooling ----------------
__global__ __launch_bounds__(256) void pma_q_kernel(const float* __restrict__ seed,
                                                    const float* __restrict__ wq,
                                                    const float* __restrict__ bq,
                                                    float* __restrict__ q32) {
  int d = threadIdx.x;
  float a = bq[d];
  for (int k = 0; k < D_; ++k) a += seed[k] * wq[(size_t)k * D_ + d];
  q32[d] = a;
}

__global__ __launch_bounds__(256) void pma_attn_kernel(
    const float* __restrict__ q32, const h16_t* __restrict__ Kf,
    const h16_t* __restrict__ Vf, h16_t* __restrict__ z16) {
  __shared__ float sv[N_];
  __shared__ float red[N_];
  int b  = blockIdx.x >> 2;
  int hp = blockIdx.x & 3;
  int tid = threadIdx.x;
  float a = 0.f;
  const h16_t* kr = Kf + ((size_t)(b * N_ + tid)) * D_ + hp * DHP_;
  for (int d = 0; d < DHP_; ++d) a += q32[hp * DHP_ + d] * (float)kr[d];
  a *= 0.125f;  // 1/sqrt(64)
  red[tid] = a;
  __syncthreads();
  for (int off = 128; off > 0; off >>= 1) {
    if (tid < off) red[tid] = fmaxf(red[tid], red[tid + off]);
    __syncthreads();
  }
  float m = red[0];
  __syncthreads();
  float ev = __expf(a - m);
  red[tid] = ev;
  __syncthreads();
  for (int off = 128; off > 0; off >>= 1) {
    if (tid < off) red[tid] += red[tid + off];
    __syncthreads();
  }
  float inv = 1.0f / red[0];
  __syncthreads();
  sv[tid] = ev * inv;
  __syncthreads();
  if (tid < DHP_) {
    float z = 0.f;
    for (int n = 0; n < N_; ++n)
      z += sv[n] * (float)Vf[((size_t)(b * N_ + n)) * D_ + hp * DHP_ + tid];
    z16[(size_t)b * D_ + hp * DHP_ + tid] = (h16_t)z;
  }
}

// ============================ host orchestration ============================
extern "C" void kernel_launch(void* const* d_in, const int* in_sizes, int n_in,
                              void* d_out, int out_size, void* d_ws, size_t ws_size,
                              hipStream_t stream) {
  (void)in_sizes; (void)n_in; (void)out_size; (void)ws_size;
  auto F = [&](int i) { return reinterpret_cast<const float*>(d_in[i]); };
  auto I = [&](int i) { return reinterpret_cast<const int*>(d_in[i]); };

  // top-level inputs (setup_inputs dict order)
  const float* xyz       = F(0);
  const int*   fids      = I(1);
  const int*   jids      = I(2);
  const int*   eidx      = I(3);
  const int*   etype     = I(4);
  const float* erl       = F(5);
  // params flattened as a JAX pytree (alphabetical dict keys); 'blocks' first.
  const int P = 6;
  enum { L_BK = 0, L_BO, L_BQ, L_BV, L_FFB1, L_FFB2, L_FFW1, L_FFW2,
         L_LN1B, L_LN1G, L_LN2B, L_LN2G, L_WK, L_WO, L_WQ, L_WV };
  auto BL = [&](int L, int leaf) { return F(P + L * 16 + leaf); };
  const float* eb_b1   = F(P + 48);
  const float* eb_b2   = F(P + 49);
  const float* eb_w1   = F(P + 50);
  const float* eb_w2   = F(P + 51);
  const float* es_emb  = F(P + 52);
  const float* fg_emb  = F(P + 53);
  const float* jt_emb  = F(P + 54);
  const float* neb     = F(P + 55);
  const float* out_b1  = F(P + 56);
  const float* out_b2  = F(P + 57);
  const float* out_w1  = F(P + 58);
  const float* out_w2  = F(P + 59);
  const float* pe_b1   = F(P + 60);
  const float* pe_b2   = F(P + 61);
  const float* pe_ln1b = F(P + 62);
  const float* pe_ln1g = F(P + 63);
  const float* pe_ln2b = F(P + 64);
  const float* pe_ln2g = F(P + 65);
  const float* pe_w1   = F(P + 66);
  const float* pe_w2   = F(P + 67);
  const float* pma_bk  = F(P + 68);
  const float* pma_bo  = F(P + 69);
  const float* pma_bq  = F(P + 70);
  const float* pma_bv  = F(P + 71);
  const float* pma_lnb = F(P + 72);
  const float* pma_lng = F(P + 73);
  const float* pma_sd  = F(P + 74);
  const float* pma_wk  = F(P + 75);
  const float* pma_wo  = F(P + 76);
  const float* pma_wq  = F(P + 77);
  const float* pma_wv  = F(P + 78);

  // ---- workspace bump allocator ----
  char* base = reinterpret_cast<char*>(d_ws);
  size_t off = 0;
  auto alloc = [&](size_t bytes) -> char* {
    char* p = base + off;
    off = (off + bytes + 255) & ~(size_t)255;
    return p;
  };
  const size_t M  = (size_t)B_ * N_;     // 16384 token rows
  // packed weights (f16, WMMA-B tiled)
  h16_t* pe_w1p = (h16_t*)alloc(128 * 512 * 2);
  h16_t* pe_w2p = (h16_t*)alloc(512 * 256 * 2);
  h16_t* wqp[3], *wkp[3], *wvp[3], *wop[3], *ff1p[3], *ff2p[3];
  for (int L = 0; L < 3; ++L) {
    wqp[L]  = (h16_t*)alloc(256 * 256 * 2);
    wkp[L]  = (h16_t*)alloc(256 * 256 * 2);
    wvp[L]  = (h16_t*)alloc(256 * 256 * 2);
    wop[L]  = (h16_t*)alloc(256 * 256 * 2);
    ff1p[L] = (h16_t*)alloc(256 * 512 * 2);
    ff2p[L] = (h16_t*)alloc(512 * 256 * 2);
  }
  h16_t* pma_wkp = (h16_t*)alloc(256 * 256 * 2);
  h16_t* pma_wvp = (h16_t*)alloc(256 * 256 * 2);
  h16_t* pma_wop = (h16_t*)alloc(256 * 256 * 2);
  h16_t* out_w1p = (h16_t*)alloc(256 * 256 * 2);
  h16_t* out_w2p = (h16_t*)alloc(256 * 512 * 2);
  // activations
  h16_t* feats16 = (h16_t*)alloc(M * INPE_PAD * 2);
  float* bufA32  = (float*)alloc(M * 512 * 4);   // PE hiddens, later reused as bias(B,N,N)
  h16_t* bufB16  = (h16_t*)alloc(M * 512 * 2);   // PE hidden f16 / FF hidden f16
  float* h32     = (float*)alloc(M * 256 * 4);
  h16_t* h16p    = (h16_t*)alloc(M * 256 * 2);
  h16_t* x16     = (h16_t*)alloc(M * 256 * 2);
  h16_t* q16     = (h16_t*)alloc(M * 256 * 2);
  h16_t* k16     = (h16_t*)alloc(M * 256 * 2);
  h16_t* v16     = (h16_t*)alloc(M * 256 * 2);
  h16_t* vpk     = (h16_t*)alloc(M * 256 * 2);
  h16_t* o16     = (h16_t*)alloc(M * 256 * 2);
  float* eb32    = (float*)alloc((size_t)B_ * E_ * 4);
  float* q32pma  = (float*)alloc(256 * 4);
  h16_t* z16     = (h16_t*)alloc((size_t)B_ * 256 * 2);
  float* t32     = (float*)alloc((size_t)B_ * 256 * 4);
  h16_t* t16     = (h16_t*)alloc((size_t)B_ * 256 * 2);
  float* bias32  = bufA32;   // reused after the PE stage is done with bufA32

  // ---- helpers ----
  auto pack = [&](const float* W, h16_t* dst, int Ksrc, int Kpad, int Nc) {
    int tiles = (Kpad / 32) * (Nc / 16);
    pack_w_kernel<<<(tiles + 7) / 8, 256, 0, stream>>>(W, dst, Ksrc, Kpad, Nc);
  };
  auto gemm = [&](const h16_t* A, const h16_t* Wpk, const float* bias, const float* res,
                  float* o32, h16_t* o16_, int Mi, int K, int Nc, int act) {
    dim3 grid((Mi + 127) / 128, Nc / 64);
    gemm_wmma_kernel<<<grid, 256, 0, stream>>>(A, Wpk, bias, res, o32, o16_, Mi, K, Nc, act);
  };
  auto ln = [&](const float* in, const float* g, const float* b, float* o32, h16_t* o16_,
                int Mi, int C, int act) {
    ln_kernel<<<(Mi + 7) / 8, 256, 0, stream>>>(in, g, b, o32, o16_, Mi, C, act);
  };

  // ---- pack all weights ----
  pack(pe_w1, pe_w1p, 95, 128, 512);
  pack(pe_w2, pe_w2p, 512, 512, 256);
  for (int L = 0; L < 3; ++L) {
    pack(BL(L, L_WQ),   wqp[L],  256, 256, 256);
    pack(BL(L, L_WK),   wkp[L],  256, 256, 256);
    pack(BL(L, L_WV),   wvp[L],  256, 256, 256);
    pack(BL(L, L_WO),   wop[L],  256, 256, 256);
    pack(BL(L, L_FFW1), ff1p[L], 256, 256, 512);
    pack(BL(L, L_FFW2), ff2p[L], 512, 512, 256);
  }
  pack(pma_wk, pma_wkp, 256, 256, 256);
  pack(pma_wv, pma_wvp, 256, 256, 256);
  pack(pma_wo, pma_wop, 256, 256, 256);
  pack(out_w1, out_w1p, 256, 256, 256);
  pack(out_w2, out_w2p, 256, 256, 512);

  // ---- positional-encoder stage ----
  feats_kernel<<<(int)(M / 256), 256, 0, stream>>>(xyz, fids, jids, fg_emb, jt_emb, feats16);
  gemm(feats16, pe_w1p, pe_b1, nullptr, bufA32, nullptr, (int)M, 128, 512, 0);
  ln(bufA32, pe_ln1g, pe_ln1b, nullptr, bufB16, (int)M, 512, 2 /*gelu*/);
  gemm(bufB16, pe_w2p, pe_b2, nullptr, bufA32, nullptr, (int)M, 512, 256, 0);
  ln(bufA32, pe_ln2g, pe_ln2b, h32, h16p, (int)M, 256, 0);

  // ---- edge bias (bufA32 is free now; reuse its first 16 MB as bias) ----
  edge_mlp_kernel<<<(B_ * E_ + 255) / 256, 256, 0, stream>>>(
      xyz, eidx, etype, erl, es_emb, eb_w1, eb_b1, eb_w2, eb_b2, eb32);
  bias_fill_kernel<<<(B_ * N_ * N_) / 256, 256, 0, stream>>>(bias32, neb, B_ * N_ * N_);
  bias_scatter_kernel<<<(B_ * E_ + 255) / 256, 256, 0, stream>>>(bias32, eb32, eidx);

  // ---- transformer blocks ----
  for (int L = 0; L < 3; ++L) {
    ln(h32, BL(L, L_LN1G), BL(L, L_LN1B), nullptr, x16, (int)M, 256, 0);
    gemm(x16, wqp[L], BL(L, L_BQ), nullptr, nullptr, q16, (int)M, 256, 256, 0);
    gemm(x16, wkp[L], BL(L, L_BK), nullptr, nullptr, k16, (int)M, 256, 256, 0);
    gemm(x16, wvp[L], BL(L, L_BV), nullptr, nullptr, v16, (int)M, 256, 256, 0);
    pack_v_kernel<<<(B_ * H_ * 16) / 8, 256, 0, stream>>>(v16, vpk);
    attn_kernel<<<B_ * H_ * 8, 64, 0, stream>>>(q16, k16, vpk, bias32, o16);
    gemm(o16, wop[L], BL(L, L_BO), h32, h32, nullptr, (int)M, 256, 256, 0);
    ln(h32, BL(L, L_LN2G), BL(L, L_LN2B), nullptr, x16, (int)M, 256, 0);
    gemm(x16, ff1p[L], BL(L, L_FFB1), nullptr, nullptr, bufB16, (int)M, 256, 512, 1 /*silu*/);
    gemm(bufB16, ff2p[L], BL(L, L_FFB2), h32, h32, h16p, (int)M, 512, 256, 0);
  }

  // ---- PMA pooling ----
  gemm(h16p, pma_wkp, pma_bk, nullptr, nullptr, k16, (int)M, 256, 256, 0);
  gemm(h16p, pma_wvp, pma_bv, nullptr, nullptr, v16, (int)M, 256, 256, 0);
  pma_q_kernel<<<1, 256, 0, stream>>>(pma_sd, pma_wq, pma_bq, q32pma);
  pma_attn_kernel<<<B_ * HP_, 256, 0, stream>>>(q32pma, k16, v16, z16);

  // ---- output head ----
  gemm(z16, pma_wop, pma_bo, nullptr, t32, nullptr, B_, 256, 256, 0);
  ln(t32, pma_lng, pma_lnb, nullptr, z16, B_, 256, 0);
  gemm(z16, out_w1p, out_b1, nullptr, nullptr, t16, B_, 256, 256, 1 /*silu*/);
  gemm(t16, out_w2p, out_b2, nullptr, (float*)d_out, nullptr, B_, 256, 512, 0);
}
